// MambaChannelAttention_58342835749267
// MI455X (gfx1250) — compile-verified, matrix operand order checked by reading, not run on hardware
//
#include <hip/hip_runtime.h>
#include <hip/hip_bf16.h>
#include <math.h>

typedef __attribute__((ext_vector_type(16))) _Float16 v16h;
typedef __attribute__((ext_vector_type(8)))  _Float16 v8h;
typedef __attribute__((ext_vector_type(8)))  float    v8f;

#define BB   8
#define CC   256
#define LL   1024
#define DII  512
#define DSS  16
#define NL   (BB * LL)      // 8192 rows

// ---------------------------------------------------------------------------
// Channel attention: pooling
// ---------------------------------------------------------------------------
__global__ void mca_pool_kernel(const float* __restrict__ x, float* __restrict__ pooled) {
    __shared__ float ssum[256];
    __shared__ float smax[256];
    int bc = blockIdx.x;                      // 0 .. B*C-1
    const float* p = x + (size_t)bc * LL;
    int t = threadIdx.x;
    float s = 0.f, m = -1e30f;
    for (int i = t; i < LL; i += 256) { float v = p[i]; s += v; m = fmaxf(m, v); }
    ssum[t] = s; smax[t] = m; __syncthreads();
    for (int off = 128; off > 0; off >>= 1) {
        if (t < off) { ssum[t] += ssum[t + off]; smax[t] = fmaxf(smax[t], smax[t + off]); }
        __syncthreads();
    }
    if (t == 0) {
        pooled[bc] = ssum[0] * (1.f / (float)LL);
        pooled[BB * CC + bc] = smax[0];
    }
}

// ---------------------------------------------------------------------------
// Channel attention: tiny shared-MLP, sigmoid(fc(avg)+fc(max))
// ---------------------------------------------------------------------------
__global__ void mca_attn_kernel(const float* __restrict__ fc1,   // (16,256)
                                const float* __restrict__ fc2,   // (256,16)
                                const float* __restrict__ pooled,
                                float* __restrict__ attn) {
    __shared__ float hidA[BB][16];
    __shared__ float hidM[BB][16];
    int t = threadIdx.x;
    if (t < BB * 16) {
        int b = t >> 4, j = t & 15;
        const float* w  = fc1 + j * CC;
        const float* pa = pooled + b * CC;
        const float* pm = pooled + BB * CC + b * CC;
        float sA = 0.f, sM = 0.f;
        for (int c = 0; c < CC; ++c) { sA += w[c] * pa[c]; sM += w[c] * pm[c]; }
        hidA[b][j] = fmaxf(sA, 0.f);
        hidM[b][j] = fmaxf(sM, 0.f);
    }
    __syncthreads();
    int c = t;  // 256 threads == 256 channels
    for (int b = 0; b < BB; ++b) {
        float aA = 0.f, aM = 0.f;
        for (int j = 0; j < 16; ++j) {
            float w = fc2[c * 16 + j];
            aA += w * hidA[b][j];
            aM += w * hidM[b][j];
        }
        float v = aA + aM;
        attn[b * CC + c] = 1.f / (1.f + __expf(-v));
    }
}

// ---------------------------------------------------------------------------
// f32 -> f16 converters
// ---------------------------------------------------------------------------
__global__ void mca_cvt_f16(const float* __restrict__ src, _Float16* __restrict__ dst, int n) {
    int i = blockIdx.x * blockDim.x + threadIdx.x;
    if (i < n) dst[i] = (_Float16)src[i];
}

// dt_proj_w (512,16) -> (512,32) zero-padded in K
__global__ void mca_cvt_dtw_pad(const float* __restrict__ src, _Float16* __restrict__ dst) {
    int i = blockIdx.x * blockDim.x + threadIdx.x;     // over 512*32
    if (i < DII * 32) {
        int k = i & 31, r = i >> 5;
        dst[i] = (k < DSS) ? (_Float16)src[r * DSS + k] : (_Float16)0.f;
    }
}

// x (B,C,L) -> xs_h (B*L, C) f16 (transpose + convert)
__global__ void mca_build_xs(const float* __restrict__ x, _Float16* __restrict__ xs_h) {
    int i = blockIdx.x * blockDim.x + threadIdx.x;     // over NL*C
    if (i >= NL * CC) return;
    int c = i & (CC - 1);
    int bl = i >> 8;
    int b = bl >> 10, l = bl & (LL - 1);
    xs_h[i] = (_Float16)x[((size_t)b * CC + c) * LL + l];
}

// dbc (NL,48) -> dtr_h (NL,32) f16, K zero-padded 16->32
__global__ void mca_extract_dtr(const float* __restrict__ dbc, _Float16* __restrict__ dtr_h) {
    int i = blockIdx.x * blockDim.x + threadIdx.x;     // over NL*32
    if (i >= NL * 32) return;
    int k = i & 31, bl = i >> 5;
    dtr_h[i] = (k < DSS) ? (_Float16)dbc[(size_t)bl * 48 + k] : (_Float16)0.f;
}

// ---------------------------------------------------------------------------
// WMMA GEMM: C(MxN,f32) = A(MxK,f16 row-major) * W(NxK,f16 row-major)^T
// One wave computes one 16x16 tile; K stepped by 32 via v_wmma_f32_16x16x32_f16.
// Fragment layout per ISA 7.12.2 (16-bit A 16x32): lanes 0-15 carry M rows with
// K = k0+[0..7] and k0+16+[0..7]; lanes 16-31 carry K = k0+8+[0..7], k0+24+[0..7].
// B fragment mirrors A over the transposed weight matrix.
// ---------------------------------------------------------------------------
__global__ void mca_wmma_gemm(const _Float16* __restrict__ A,
                              const _Float16* __restrict__ W,
                              float* __restrict__ C,
                              int M, int N, int K) {
    int wave = (blockIdx.x * blockDim.x + threadIdx.x) >> 5;
    int lane = threadIdx.x & 31;
    int ntn  = N >> 4;
    int total = (M >> 4) * ntn;
    if (wave >= total) return;                 // whole-wave uniform: EXEC stays all-1s
    int tm = (wave / ntn) << 4;
    int tn = (wave % ntn) << 4;
    int row = lane & 15;
    int hi  = lane >> 4;                       // half-wave select

    v8f acc = {};
    const _Float16* arow = A + (size_t)(tm + row) * K + hi * 8;
    const _Float16* brow = W + (size_t)(tn + row) * K + hi * 8;
    for (int k0 = 0; k0 < K; k0 += 32) {
        union { v16h v; v8h h[2]; } af, bf;
        af.h[0] = *(const v8h*)(arow + k0);
        af.h[1] = *(const v8h*)(arow + k0 + 16);
        bf.h[0] = *(const v8h*)(brow + k0);
        bf.h[1] = *(const v8h*)(brow + k0 + 16);
        acc = __builtin_amdgcn_wmma_f32_16x16x32_f16(
            false, af.v, false, bf.v, (short)0, acc, false, false);
    }
    // D layout: lane<16 -> rows tm+0..7, lane>=16 -> rows tm+8..15, col tn+row
    float* cp = C + (size_t)(tm + hi * 8) * N + tn + row;
    #pragma unroll
    for (int v = 0; v < 8; ++v) cp[(size_t)v * N] = acc[v];
}

// ---------------------------------------------------------------------------
// Depthwise causal conv (K=4, left pad 3) + bias + silu. Emits f32 and f16.
// ---------------------------------------------------------------------------
__global__ void mca_conv_silu(const float* __restrict__ xz,   // (NL, 2*DI)
                              const float* __restrict__ cw,   // (DI,1,4)
                              const float* __restrict__ cb,   // (DI,)
                              float* __restrict__ xc,
                              _Float16* __restrict__ xc_h) {
    int i = blockIdx.x * blockDim.x + threadIdx.x;            // over NL*DI
    if (i >= NL * DII) return;
    int d = i & (DII - 1);
    int bl = i >> 9;
    int b = bl >> 10, l = bl & (LL - 1);
    float acc = cb[d];
    #pragma unroll
    for (int k = 0; k < 4; ++k) {
        int lk = l + k - 3;
        if (lk >= 0) acc += cw[d * 4 + k] * xz[((size_t)(b * LL + lk)) * (2 * DII) + d];
    }
    float s = acc / (1.f + __expf(-acc));                     // silu
    xc[i] = s;
    xc_h[i] = (_Float16)s;
}

// softplus(dt_lin + dt_b), stable logaddexp form (in place)
__global__ void mca_softplus(float* __restrict__ dt, const float* __restrict__ dtb) {
    int i = blockIdx.x * blockDim.x + threadIdx.x;            // over NL*DI
    if (i >= NL * DII) return;
    float v = dt[i] + dtb[i & (DII - 1)];
    dt[i] = fmaxf(v, 0.f) + log1pf(__expf(-fabsf(v)));
}

// ---------------------------------------------------------------------------
// Selective scan: thread = (b, d, s); 16-lane shfl_xor butterfly reduces h*C.
// Fuses +xc*D, *silu(z); writes y as f16 for the out_proj WMMA GEMM.
// ---------------------------------------------------------------------------
__global__ void mca_scan(const float* __restrict__ dt,    // (NL, DI)
                         const float* __restrict__ xc,    // (NL, DI)
                         const float* __restrict__ dbc,   // (NL, 48)
                         const float* __restrict__ xz,    // (NL, 2*DI)  (z half)
                         const float* __restrict__ A_log, // (DI, DS)
                         const float* __restrict__ Dp,    // (DI,)
                         _Float16* __restrict__ y_h) {    // (NL, DI)
    int t = blockIdx.x * blockDim.x + threadIdx.x;        // B*DI*DS = 65536
    int s = t & 15;
    int g = t >> 4;                                       // b*DI + d
    int d = g & (DII - 1);
    int b = g >> 9;
    float A  = -__expf(A_log[d * DSS + s]);
    float Dd = Dp[d];
    float h = 0.f;
    for (int l = 0; l < LL; ++l) {
        size_t base = (size_t)(b * LL + l);
        float dtv = dt[base * DII + d];
        float xcv = xc[base * DII + d];
        float Bv  = dbc[base * 48 + 16 + s];
        float Cv  = dbc[base * 48 + 32 + s];
        float dA  = __expf(dtv * A);
        h = dA * h + dtv * Bv * xcv;
        float p = h * Cv;
        p += __shfl_xor(p, 8);
        p += __shfl_xor(p, 4);
        p += __shfl_xor(p, 2);
        p += __shfl_xor(p, 1);
        if (s == 0) {
            float zv  = xz[base * (2 * DII) + DII + d];
            float sig = 1.f / (1.f + __expf(-zv));
            float yv  = (p + xcv * Dd) * (zv * sig);
            y_h[base * DII + d] = (_Float16)yv;
        }
    }
}

// out = x * attn + 0.1 * mamba (mamba in (B*L, C))
__global__ void mca_final(const float* __restrict__ x,
                          const float* __restrict__ attn,
                          const float* __restrict__ mamba,
                          float* __restrict__ out) {
    int i = blockIdx.x * blockDim.x + threadIdx.x;        // over B*C*L
    if (i >= BB * CC * LL) return;
    int l = i & (LL - 1);
    int bc = i >> 10;
    int b = bc >> 8;
    int c = bc & (CC - 1);
    out[i] = x[i] * attn[bc] + 0.1f * mamba[((size_t)(b * LL + l)) * CC + c];
}

// ---------------------------------------------------------------------------
extern "C" void kernel_launch(void* const* d_in, const int* in_sizes, int n_in,
                              void* d_out, int out_size, void* d_ws, size_t ws_size,
                              hipStream_t stream) {
    (void)in_sizes; (void)n_in; (void)out_size; (void)ws_size;
    const float* x      = (const float*)d_in[0];
    const float* fc1_w  = (const float*)d_in[1];
    const float* fc2_w  = (const float*)d_in[2];
    const float* inp_w  = (const float*)d_in[3];
    const float* conv_w = (const float*)d_in[4];
    const float* conv_b = (const float*)d_in[5];
    const float* xp_w   = (const float*)d_in[6];
    const float* dt_w   = (const float*)d_in[7];
    const float* dt_b   = (const float*)d_in[8];
    const float* A_log  = (const float*)d_in[9];
    const float* Dp     = (const float*)d_in[10];
    const float* out_w  = (const float*)d_in[11];
    float* out = (float*)d_out;

    char* p = (char*)d_ws;
    auto alloc = [&](size_t bytes) -> void* {
        void* r = (void*)p;
        p += (bytes + 255) & ~(size_t)255;
        return r;
    };
    _Float16* xs_h  = (_Float16*)alloc((size_t)NL * CC * 2);        // A for in_proj
    _Float16* inw_h = (_Float16*)alloc((size_t)2 * DII * CC * 2);   // (1024,256)
    _Float16* xpw_h = (_Float16*)alloc((size_t)48 * DII * 2);       // (48,512)
    _Float16* dtw_h = (_Float16*)alloc((size_t)DII * 32 * 2);       // (512,32) padded
    _Float16* ow_h  = (_Float16*)alloc((size_t)CC * DII * 2);       // (256,512)
    float*    xz    = (float*)alloc((size_t)NL * 2 * DII * 4);      // (8192,1024)
    float*    xc    = (float*)alloc((size_t)NL * DII * 4);
    _Float16* xc_h  = (_Float16*)alloc((size_t)NL * DII * 2);
    float*    dbc   = (float*)alloc((size_t)NL * 48 * 4);
    _Float16* dtr_h = (_Float16*)alloc((size_t)NL * 32 * 2);
    float*    dt    = (float*)alloc((size_t)NL * DII * 4);
    _Float16* y_h   = (_Float16*)alloc((size_t)NL * DII * 2);
    float*    mamba = (float*)alloc((size_t)NL * CC * 4);
    float*    pooled= (float*)alloc((size_t)2 * BB * CC * 4);
    float*    attn  = (float*)alloc((size_t)BB * CC * 4);

    auto gemm_blocks = [](int M, int N) {
        long waves = (long)(M >> 4) * (N >> 4);
        return (int)((waves * 32 + 255) / 256);
    };

    // --- channel attention ---
    mca_pool_kernel<<<BB * CC, 256, 0, stream>>>(x, pooled);
    mca_attn_kernel<<<1, 256, 0, stream>>>(fc1_w, fc2_w, pooled, attn);

    // --- weight conversion to f16 ---
    mca_cvt_f16<<<(2 * DII * CC + 255) / 256, 256, 0, stream>>>(inp_w, inw_h, 2 * DII * CC);
    mca_cvt_f16<<<(48 * DII + 255) / 256, 256, 0, stream>>>(xp_w, xpw_h, 48 * DII);
    mca_cvt_f16<<<(CC * DII + 255) / 256, 256, 0, stream>>>(out_w, ow_h, CC * DII);
    mca_cvt_dtw_pad<<<(DII * 32 + 255) / 256, 256, 0, stream>>>(dt_w, dtw_h);

    // --- in_proj: xz = xs @ in_proj_w^T  (M=8192, N=1024, K=256) ---
    mca_build_xs<<<(NL * CC) / 256, 256, 0, stream>>>(x, xs_h);
    mca_wmma_gemm<<<gemm_blocks(NL, 2 * DII), 256, 0, stream>>>(xs_h, inw_h, xz, NL, 2 * DII, CC);

    // --- depthwise causal conv + silu ---
    mca_conv_silu<<<(NL * DII) / 256, 256, 0, stream>>>(xz, conv_w, conv_b, xc, xc_h);

    // --- x_proj: dbc = xc @ x_proj_w^T  (M=8192, N=48, K=512) ---
    mca_wmma_gemm<<<gemm_blocks(NL, 48), 256, 0, stream>>>(xc_h, xpw_h, dbc, NL, 48, DII);

    // --- dt_proj: dt_lin = dt_r @ dt_proj_w^T  (M=8192, N=512, K=32 padded) ---
    mca_extract_dtr<<<(NL * 32) / 256, 256, 0, stream>>>(dbc, dtr_h);
    mca_wmma_gemm<<<gemm_blocks(NL, DII), 256, 0, stream>>>(dtr_h, dtw_h, dt, NL, DII, 32);
    mca_softplus<<<(NL * DII) / 256, 256, 0, stream>>>(dt, dt_b);

    // --- selective scan (fused D-skip + silu(z) gate), emits f16 ---
    mca_scan<<<(BB * DII * DSS) / 256, 256, 0, stream>>>(dt, xc, dbc, xz, A_log, Dp, y_h);

    // --- out_proj: mamba = y @ out_proj_w^T  (M=8192, N=256, K=512) ---
    mca_wmma_gemm<<<gemm_blocks(NL, CC), 256, 0, stream>>>(y_h, ow_h, mamba, NL, CC, DII);

    // --- final combine ---
    mca_final<<<(BB * CC * LL) / 256, 256, 0, stream>>>(x, attn, mamba, out);
}
